// SparseAttLayer_3607772529169
// MI455X (gfx1250) — matrix operand order repaired and checked by reading, not compiled, not run on hardware
//
#include <hip/hip_runtime.h>

// ---------------------------------------------------------------------------
// SparseAttLayer fused kernel for MI455X (gfx1250, wave32, WMMA + async DMA).
//
//   att[b,k,o,f] = sum_e x[b,f,e] * W[k,o,e],  W[k,o,e] = sum_y bil[k,e,y]*Q[k,o,y]
//   out[b,k,o,f] = entmax15(att over f) * values[k,o,f]
//
// Shapes: B=2048, K=8, F=128, E=128, O=64.
// ---------------------------------------------------------------------------

typedef __bf16 bf16_t;
typedef __attribute__((ext_vector_type(16))) __bf16 v16bf;
typedef __attribute__((ext_vector_type(8)))  float  v8f;

#define K_H   8
#define F_D   128
#define E_D   128
#define O_D   64
#define NITER 50

#define LDX 136          // xs row stride (bf16): 272 B rows, 16B-aligned chunks
#define LDW 136          // wl row stride (bf16): 272 B rows, 16B-aligned chunks
#define LDA 132          // att row stride (f32): float4-aligned, bank-skewed
#define DM0 0.9116116523516815f   // 1 - (1/128)^(alpha-1), alpha = 1.5

// ---------------------------------------------------------------------------
// Kernel 1: W[k][o][e] = sum_y bilinear[k][e][y] * Q[k][o][y]  (bf16 to ws)
// ---------------------------------------------------------------------------
__global__ __launch_bounds__(256) void prep_w_kernel(
    const float* __restrict__ Q, const float* __restrict__ bil,
    bf16_t* __restrict__ w2)
{
    const int k   = blockIdx.x;
    const int idx = blockIdx.y * 256 + threadIdx.x;   // 0 .. O*E-1
    const int o   = idx >> 7;
    const int e   = idx & 127;
    const float* __restrict__ qr = Q   + ((size_t)k * O_D + o) * E_D;
    const float* __restrict__ br = bil + ((size_t)k * E_D + e) * E_D;
    float s = 0.0f;
#pragma unroll 8
    for (int y = 0; y < E_D; ++y) s = fmaf(qr[y], br[y], s);
    w2[(size_t)k * (O_D * E_D) + idx] = (bf16_t)s;
}

// ---------------------------------------------------------------------------
// Async DMA of one head's W tile (64x128 bf16, row-major) into padded LDS,
// via CDNA5 global_load_async_to_lds_b128 (tracked with ASYNCcnt).
// Each thread issues 4 x 16B chunks; waited with s_wait_asynccnt at k-top.
// ---------------------------------------------------------------------------
__device__ __forceinline__ void issue_wl_async(const bf16_t* __restrict__ w2k,
                                               bf16_t* wl, int t)
{
    const unsigned long long gbase = (unsigned long long)(size_t)w2k;
#pragma unroll
    for (int i = 0; i < 4; ++i) {
        const int c = t + i * 256;            // 1024 chunks of 8 halfs
        const int o = c >> 4;                 // 16 chunks per 128-elem row
        const int p = c & 15;
        // AS(3) offset == low 32 bits of the generic pointer to __shared__
        const unsigned lds  = (unsigned)(size_t)(&wl[o * LDW + p * 8]);
        const unsigned goff = (unsigned)((o * E_D + p * 8) * (int)sizeof(bf16_t));
        asm volatile("global_load_async_to_lds_b128 %0, %1, %2"
                     :: "v"(lds), "v"(goff), "s"(gbase) : "memory");
    }
}

__device__ __forceinline__ void wait_async_zero() {
    asm volatile("s_wait_asynccnt 0x0" ::: "memory");
}

// ---------------------------------------------------------------------------
// Kernel 2: per-b fused GEMM (bf16 WMMA) + entmax bisection + scale by values.
// ---------------------------------------------------------------------------
__global__ __launch_bounds__(256) void fused_kernel(
    const float* __restrict__ x, const bf16_t* __restrict__ w2,
    const float* __restrict__ values, float* __restrict__ out)
{
    __shared__ bf16_t xs[F_D * LDX];      // x[b] transcoded to bf16  (34.0 KB)
    __shared__ bf16_t wl[O_D * LDW];      // current head's W tile    (17.0 KB)
    __shared__ float  att[32 * LDA];      // one 32-row half of att   (16.5 KB)

    const int b    = blockIdx.x;
    const int t    = threadIdx.x;
    const int lane = t & 31;
    const int wave = t >> 5;
    const int lm   = lane & 15;           // M/N index within a 16-wide tile
    const int hi   = lane >> 4;           // which K/M half this lane holds

    // Kick off W(k=0) DMA first: overlaps the x f32->bf16 staging below.
    issue_wl_async(w2, wl, t);

    // ---- stage 0: load x[b] (f32, coalesced b128) -> xs (bf16) -------------
    {
        const float4* __restrict__ xb = (const float4*)(x + (size_t)b * (F_D * E_D));
#pragma unroll
        for (int i = 0; i < 16; ++i) {
            int c4  = t + i * 256;            // 4096 float4 chunks
            int row = c4 >> 5;                // 32 chunks per 128-elem row
            int col = (c4 & 31) << 2;
            float4 v = xb[c4];
            bf16_t* d = &xs[row * LDX + col];
            d[0] = (bf16_t)v.x; d[1] = (bf16_t)v.y;
            d[2] = (bf16_t)v.z; d[3] = (bf16_t)v.w;
        }
    }

    // GEMM tile ownership: 8 waves cover a 32(O) x 128(F) half-tile.
    const int mt  = wave >> 2;            // M-tile within half: 0..1  (16 rows each)
    const int nt0 = (wave & 3) * 2;       // two N-tiles: nt0, nt0+1   (16 cols each)

    for (int k = 0; k < K_H; ++k) {
        // Warm L2 for next head's values while this head computes.
        if (k + 1 < K_H)
            __builtin_prefetch(values + (size_t)(k + 1) * (O_D * F_D) + t * 32, 0, 2);

        wait_async_zero();     // this wave's W DMA chunks have landed in LDS
        __syncthreads();       // all waves' chunks visible (and xs on k==0)

        for (int h = 0; h < 2; ++h) {
            const int obase = h * 32 + mt * 16;   // O row of this wave's A tile

            // ---- WMMA GEMM: D[o,f] += W[o,e] * xT[e,f], e = 0..127 ---------
            // Inner loop is pure LDS + WMMA: no VMEM on the critical path.
            v8f acc0 = {};
            v8f acc1 = {};
#pragma unroll
            for (int es = 0; es < 4; ++es) {
                const int eb = es * 32;

                // A fragment (16x32 bf16): lane holds row o=obase+lm,
                // K = eb + hi*16 + {0..15}.
                v16bf a;
                {
                    const bf16_t* ar = &wl[(obase + lm) * LDW + eb + hi * 16];
                    ((uint4*)&a)[0] = *(const uint4*)(ar);
                    ((uint4*)&a)[1] = *(const uint4*)(ar + 8);
                }
                // B fragments (32x16 bf16): lane holds col f, K = eb+16*hi+{0..15}.
                v16bf b0, b1;
                {
                    const bf16_t* br0 = &xs[(nt0 * 16 + lm) * LDX + eb + hi * 16];
                    const bf16_t* br1 = &xs[((nt0 + 1) * 16 + lm) * LDX + eb + hi * 16];
                    ((uint4*)&b0)[0] = *(const uint4*)(br0);
                    ((uint4*)&b0)[1] = *(const uint4*)(br0 + 8);
                    ((uint4*)&b1)[0] = *(const uint4*)(br1);
                    ((uint4*)&b1)[1] = *(const uint4*)(br1 + 8);
                }
                acc0 = __builtin_amdgcn_wmma_f32_16x16x32_bf16(
                           false, a, false, b0, (short)0, acc0, false, false);
                acc1 = __builtin_amdgcn_wmma_f32_16x16x32_bf16(
                           false, a, false, b1, (short)0, acc1, false, false);
            }

            __syncthreads();   // all GEMM reads of wl/att for this phase done

            // wl fully consumed for this head after h==1's GEMM: start next
            // head's DMA now so it overlaps att stores + the entmax loop.
            if (h == 1 && k + 1 < K_H)
                issue_wl_async(w2 + (size_t)(k + 1) * (O_D * E_D), wl, t);

            // D layout: VGPR r -> M = r + 8*hi, N = lm.
            {
                const int rloc0 = mt * 16 + 8 * hi;      // local row base in half
#pragma unroll
                for (int r = 0; r < 8; ++r) {
                    att[(rloc0 + r) * LDA + nt0 * 16 + lm]       = acc0[r];
                    att[(rloc0 + r) * LDA + (nt0 + 1) * 16 + lm] = acc1[r];
                }
            }
            __syncthreads();   // att half-tile ready

            // ---- entmax-1.5 bisection over f (128), exact reference math ---
            // 8 lanes per row, 16 values per lane -> 3-step xor reduction.
            const int rloc = wave * 4 + (lane >> 3);   // local row 0..31
            const int o    = h * 32 + rloc;            // global O index
            const int fb   = (lane & 7) * 16;          // this lane's f base

            float Xs[16];
            {
                const float* ap = &att[rloc * LDA + fb];
#pragma unroll
                for (int j4 = 0; j4 < 4; ++j4) {
                    float4 v = *(const float4*)(ap + 4 * j4);
                    Xs[4 * j4 + 0] = v.x * 0.5f;   // * (alpha - 1)
                    Xs[4 * j4 + 1] = v.y * 0.5f;
                    Xs[4 * j4 + 2] = v.z * 0.5f;
                    Xs[4 * j4 + 3] = v.w * 0.5f;
                }
            }

            float mx = Xs[0];
#pragma unroll
            for (int j = 1; j < 16; ++j) mx = fmaxf(mx, Xs[j]);
            mx = fmaxf(mx, __shfl_xor(mx, 1, 32));
            mx = fmaxf(mx, __shfl_xor(mx, 2, 32));
            mx = fmaxf(mx, __shfl_xor(mx, 4, 32));

            float tau_lo = mx - 1.0f;
            float s = 0.0f;
#pragma unroll
            for (int j = 0; j < 16; ++j) {
                float tv = fmaxf(Xs[j] - tau_lo, 0.0f);
                s = fmaf(tv, tv, s);
            }
            s += __shfl_xor(s, 1, 32);
            s += __shfl_xor(s, 2, 32);
            s += __shfl_xor(s, 4, 32);
            const float f_lo = s - 1.0f;

            float dm    = DM0;
            float tau_m = tau_lo + DM0;
            for (int it = 0; it < NITER; ++it) {
                dm *= 0.5f;
                tau_m = tau_lo + dm;
                float sm = 0.0f;
#pragma unroll
                for (int j = 0; j < 16; ++j) {
                    float tv = fmaxf(Xs[j] - tau_m, 0.0f);
                    sm = fmaf(tv, tv, sm);
                }
                sm += __shfl_xor(sm, 1, 32);
                sm += __shfl_xor(sm, 2, 32);
                sm += __shfl_xor(sm, 4, 32);
                if ((sm - 1.0f) * f_lo >= 0.0f) tau_lo = tau_m;
            }

            float p[16];
            float ss = 0.0f;
#pragma unroll
            for (int j = 0; j < 16; ++j) {
                float tv = fmaxf(Xs[j] - tau_m, 0.0f);
                p[j] = tv * tv;
                ss += p[j];
            }
            ss += __shfl_xor(ss, 1, 32);
            ss += __shfl_xor(ss, 2, 32);
            ss += __shfl_xor(ss, 4, 32);
            const float inv = 1.0f / ss;

            // ---- scale by values[k,o,f] and store (b128, lane-contiguous) --
            const float* __restrict__ vp =
                values + ((size_t)k * O_D + o) * F_D + fb;
            float* __restrict__ op =
                out + ((((size_t)b * K_H + k) * O_D + o) * (size_t)F_D) + fb;
#pragma unroll
            for (int j4 = 0; j4 < 4; ++j4) {
                float4 vv = *(const float4*)(vp + 4 * j4);
                float4 r;
                r.x = p[4 * j4 + 0] * inv * vv.x;
                r.y = p[4 * j4 + 1] * inv * vv.y;
                r.z = p[4 * j4 + 2] * inv * vv.z;
                r.w = p[4 * j4 + 3] * inv * vv.w;
                *(float4*)(op + 4 * j4) = r;
            }
        } // h
    } // k
}

// ---------------------------------------------------------------------------
extern "C" void kernel_launch(void* const* d_in, const int* in_sizes, int n_in,
                              void* d_out, int out_size, void* d_ws, size_t ws_size,
                              hipStream_t stream) {
    (void)n_in; (void)out_size; (void)ws_size;
    const float* x      = (const float*)d_in[0];
    const float* Q      = (const float*)d_in[1];
    const float* bil    = (const float*)d_in[2];
    const float* values = (const float*)d_in[3];
    float*  out = (float*)d_out;
    bf16_t* w2  = (bf16_t*)d_ws;          // K*O*E bf16 = 128 KB scratch

    const int B = in_sizes[0] / (F_D * E_D);

    prep_w_kernel<<<dim3(K_H, (O_D * E_D) / 256), 256, 0, stream>>>(Q, bil, w2);
    fused_kernel<<<dim3(B), 256, 0, stream>>>(x, w2, values, out);
}
